// RSNN_FWP_71760313582243
// MI455X (gfx1250) — matrix operand order repaired and checked by reading, not compiled, not run on hardware
//
#include <hip/hip_runtime.h>
#include <hip/hip_bf16.h>
#include <math.h>

typedef __attribute__((ext_vector_type(16))) _Float16 v16h;
typedef __attribute__((ext_vector_type(8)))  float    v8f;

#define B_    256
#define T_    64
#define NIN   700
#define NINP  704          // pad K to multiple of 32
#define H_    2048
#define O_    20
#define MT    (B_*T_)      // 16384 rows of the input-projection GEMM
#define KT_IN (NINP/32)    // 22 k-tiles
#define KT_HH (H_/32)      // 64 k-tiles
#define THRESH_ 0.3f

// ---- workspace layout (bytes) ----
#define OFF_AIN   ((size_t)0)
#define SZ_AIN    ((size_t)MT*NINP*2)         // swizzled f16 input (A layout)
#define OFF_WIH   (OFF_AIN + SZ_AIN)
#define SZ_WIH    ((size_t)H_*NINP*2)         // swizzled f16 W_ih (B layout)
#define OFF_WHH   (OFF_WIH + SZ_WIH)
#define SZ_WHH    ((size_t)H_*H_*2)           // swizzled f16 W_hh (B layout)
#define OFF_XPROJ (OFF_WHH + SZ_WHH)
#define SZ_XPROJ  ((size_t)T_*B_*H_*4)        // f32 x_proj[t][b][h] (+bias)
#define OFF_HMEM  (OFF_XPROJ + SZ_XPROJ)
#define SZ_HMEM   ((size_t)B_*H_*4)
#define OFF_SNAT  (OFF_HMEM + SZ_HMEM)        // spikes, natural f32 [B,H]
#define SZ_SNAT   ((size_t)B_*H_*4)
#define OFF_SSW0  (OFF_SNAT + SZ_SNAT)        // spikes, swizzled f16 A-layout, buf 0
#define SZ_SSW    ((size_t)B_*H_*2)
#define OFF_SSW1  (OFF_SSW0 + SZ_SSW)         // buf 1 (double buffered across steps)
#define OFF_OMEM  (OFF_SSW1 + SZ_SSW)
#define SZ_OV     ((size_t)B_*O_*4)
#define OFF_OSPK  (OFF_OMEM + SZ_OV)
#define OFF_OSUM  (OFF_OSPK + SZ_OV)
#define ZERO_BYTES (SZ_HMEM + SZ_SNAT + 2*SZ_SSW + 3*SZ_OV)

static __device__ __forceinline__ v8f wmma_f16(v16h a, v16h b, v8f c) {
    return __builtin_amdgcn_wmma_f32_16x16x32_f16(false, a, false, b, (short)0, c,
                                                  false, false);
}

// zero the recurrent-state region of the workspace
__global__ void zero_ws(unsigned int* __restrict__ p, size_t n) {
    size_t i = (size_t)blockIdx.x * blockDim.x + threadIdx.x;
    if (i < n) p[i] = 0u;
}

// f32 row-major [rows, ksrc] -> f16 WMMA A-operand tiles (16 x 32), zero-padded K
// A layout: lane m=L%16; lanes<16: K {0..7,16..23}; lanes>=16: K {8..15,24..31}
__global__ void swz_a(const float* __restrict__ src, _Float16* __restrict__ dst,
                      int rows, int ksrc, int ktiles) {
    int tid  = blockIdx.x * blockDim.x + threadIdx.x;
    int lane = tid & 31;
    int tile = tid >> 5;
    if (tile >= (rows / 16) * ktiles) return;
    int mt = tile / ktiles, kt = tile % ktiles;
    int m = mt * 16 + (lane & 15);
    int kofs = (lane >= 16) ? 8 : 0;
    v16h out;
#pragma unroll
    for (int v = 0; v < 8; ++v) {
#pragma unroll
        for (int p = 0; p < 2; ++p) {
            int kl = ((v < 4) ? (2 * v + p) : (16 + 2 * (v - 4) + p)) + kofs;
            int k  = kt * 32 + kl;
            float x = (k < ksrc) ? src[(size_t)m * ksrc + k] : 0.0f;
            out[v * 2 + p] = (_Float16)x;
        }
    }
    *(v16h*)(dst + (size_t)tile * 512 + lane * 16) = out;
}

// f32 row-major weight W[n, k] -> f16 WMMA B-operand tiles (B[k][n] = W[n][k])
// B layout: lane n=L%16; lanes<16: K {0..7,16..23}; lanes>=16: K {8..15,24..31} (2 per VGPR)
__global__ void swz_b(const float* __restrict__ w, _Float16* __restrict__ dst,
                      int nrows, int ksrc, int ktiles) {
    int tid  = blockIdx.x * blockDim.x + threadIdx.x;
    int lane = tid & 31;
    int tile = tid >> 5;
    if (tile >= (nrows / 16) * ktiles) return;
    int nt = tile / ktiles, kt = tile % ktiles;
    int n = nt * 16 + (lane & 15);
    int kofs = (lane >= 16) ? 8 : 0;
    v16h out;
#pragma unroll
    for (int v = 0; v < 8; ++v) {
#pragma unroll
        for (int p = 0; p < 2; ++p) {
            int k = kt * 32 + kofs + ((v < 4) ? (2 * v + p) : (16 + 2 * (v - 4) + p));
            float x = (k < ksrc) ? w[(size_t)n * ksrc + k] : 0.0f;
            out[v * 2 + p] = (_Float16)x;
        }
    }
    *(v16h*)(dst + (size_t)tile * 512 + lane * 16) = out;
}

// x_proj[t][b][h] = input[b,t,:] . W_ih[h,:] + b_ih[h]
// 2x2 register blocking: each wave computes a 32(M) x 32(N) block = 4 C tiles,
// sharing 2 A-tile loads + 2 B-tile loads per K-tile (halves L2 traffic).
__global__ __launch_bounds__(256)
void xproj_gemm(const _Float16* __restrict__ a_sw, const _Float16* __restrict__ b_sw,
                const float* __restrict__ b_ih, float* __restrict__ xproj) {
    int lane = threadIdx.x & 31;
    int wid  = threadIdx.x >> 5;
    int task = blockIdx.x * 8 + wid;       // (MT/32)*(H_/32) = 512*64 = 32768 tasks
    int mtp = task >> 6;                   // 0..511  (32-row block)
    int ntp = task & 63;                   // 0..63   (32-col block)
    int mt0 = 2 * mtp, mt1 = mt0 + 1;

    v8f acc00 = {}, acc01 = {}, acc10 = {}, acc11 = {};
    const v16h* ap0 = (const v16h*)a_sw + (size_t)mt0 * KT_IN * 32 + lane;
    const v16h* ap1 = (const v16h*)a_sw + (size_t)mt1 * KT_IN * 32 + lane;
    const v16h* bp0 = (const v16h*)b_sw + (size_t)(2 * ntp) * KT_IN * 32 + lane;
    const v16h* bp1 = bp0 + (size_t)KT_IN * 32;
    for (int kt = 0; kt < KT_IN; ++kt) {
        v16h a0 = ap0[kt * 32];
        v16h a1 = ap1[kt * 32];
        v16h b0 = bp0[kt * 32];
        v16h b1 = bp1[kt * 32];
        acc00 = wmma_f16(a0, b0, acc00);
        acc01 = wmma_f16(a0, b1, acc01);
        acc10 = wmma_f16(a1, b0, acc10);
        acc11 = wmma_f16(a1, b1, acc11);
    }
    int n_loc = lane & 15;
    int mbase = (lane >= 16) ? 8 : 0;
    int col0 = ntp * 32 + n_loc;
    int col1 = col0 + 16;
    float bias0 = b_ih[col0], bias1 = b_ih[col1];
#pragma unroll
    for (int r = 0; r < 8; ++r) {
        int row0 = mt0 * 16 + mbase + r;   // row = b*T + t
        int row1 = row0 + 16;
        int bb0 = row0 >> 6, tt0 = row0 & 63;
        int bb1 = row1 >> 6, tt1 = row1 & 63;
        xproj[((size_t)tt0 * B_ + bb0) * H_ + col0] = acc00[r] + bias0;
        xproj[((size_t)tt0 * B_ + bb0) * H_ + col1] = acc01[r] + bias1;
        xproj[((size_t)tt1 * B_ + bb1) * H_ + col0] = acc10[r] + bias0;
        xproj[((size_t)tt1 * B_ + bb1) * H_ + col1] = acc11[r] + bias1;
    }
}

// scatter one spike into the swizzled A-layout buffer
static __device__ __forceinline__ void scat_spike(_Float16* __restrict__ wtile,
                                                  int k, int m_loc, float s) {
    int la  = m_loc + (((k & 15) >= 8) ? 16 : 0);
    int idx = (k & 7) + ((k >= 16) ? 8 : 0);
    wtile[la * 16 + idx] = (_Float16)s;
}

// One recurrence step for the hidden layer. 2x2 register blocking:
// each wave computes 32(M) x 32(N) = 4 WMMA C tiles from 2 A + 2 B loads per
// K-tile (halves the per-step L2 read traffic on L2-resident W_hh).
// Fused LIF epilogue + spike writeback into BOTH the natural f32 buffer and
// the next step's swizzled f16 A buffer (wave owns A k-tiles (mt0,ntp),(mt1,ntp)).
__global__ __launch_bounds__(256)
void step_hidden(const _Float16* __restrict__ whh_sw, const _Float16* __restrict__ s_rd,
                 _Float16* __restrict__ s_wr, float* __restrict__ h_mem,
                 float* __restrict__ s_nat, const float* __restrict__ xproj_t,
                 const float* __restrict__ b_hh, const float* __restrict__ tau_h) {
    int lane = threadIdx.x & 31;
    int wid  = threadIdx.x >> 5;
    int task = blockIdx.x * 8 + wid;       // 8 * 64 = 512 tasks exactly
    int mtp = task >> 6;                   // batch 32-row block 0..7
    int ntp = task & 63;                   // hidden 32-col block 0..63
    int mt0 = 2 * mtp, mt1 = mt0 + 1;
    float alpha = __expf(-1.0f / tau_h[0]);

    v8f acc00 = {}, acc01 = {}, acc10 = {}, acc11 = {};
    const v16h* ap0 = (const v16h*)s_rd + (size_t)mt0 * KT_HH * 32 + lane;
    const v16h* ap1 = (const v16h*)s_rd + (size_t)mt1 * KT_HH * 32 + lane;
    const v16h* bp0 = (const v16h*)whh_sw + (size_t)(2 * ntp) * KT_HH * 32 + lane;
    const v16h* bp1 = bp0 + (size_t)KT_HH * 32;
    for (int kt = 0; kt < KT_HH; ++kt) {
        v16h a0 = ap0[kt * 32];
        v16h a1 = ap1[kt * 32];
        v16h b0 = bp0[kt * 32];
        v16h b1 = bp1[kt * 32];
        acc00 = wmma_f16(a0, b0, acc00);
        acc01 = wmma_f16(a0, b1, acc01);
        acc10 = wmma_f16(a1, b0, acc10);
        acc11 = wmma_f16(a1, b1, acc11);
    }

    int n_loc = lane & 15;
    int mbase = (lane >= 16) ? 8 : 0;
    int col0 = ntp * 32 + n_loc;           // columns of left C tiles
    int col1 = col0 + 16;                  // columns of right C tiles
    float bias0 = b_hh[col0], bias1 = b_hh[col1];
    // this wave owns two A k-tiles of the next-step spike buffer
    _Float16* wtile0 = s_wr + ((size_t)mt0 * KT_HH + ntp) * 512;
    _Float16* wtile1 = s_wr + ((size_t)mt1 * KT_HH + ntp) * 512;
    int k0 = col0 & 31;                    // = n_loc      (0..15)
    int k1 = col1 & 31;                    // = n_loc + 16 (16..31)

#pragma unroll
    for (int r = 0; r < 8; ++r) {
        int m_loc = mbase + r;
        int b0_idx = mt0 * 16 + m_loc;
        int b1_idx = mt1 * 16 + m_loc;
        size_t i00 = (size_t)b0_idx * H_ + col0;
        size_t i01 = (size_t)b0_idx * H_ + col1;
        size_t i10 = (size_t)b1_idx * H_ + col0;
        size_t i11 = (size_t)b1_idx * H_ + col1;

        float hm00 = h_mem[i00] * alpha * (1.0f - s_nat[i00]) + xproj_t[i00] + acc00[r] + bias0;
        float hm01 = h_mem[i01] * alpha * (1.0f - s_nat[i01]) + xproj_t[i01] + acc01[r] + bias1;
        float hm10 = h_mem[i10] * alpha * (1.0f - s_nat[i10]) + xproj_t[i10] + acc10[r] + bias0;
        float hm11 = h_mem[i11] * alpha * (1.0f - s_nat[i11]) + xproj_t[i11] + acc11[r] + bias1;

        float s00 = ((hm00 - THRESH_) > 0.0f) ? 1.0f : 0.0f;
        float s01 = ((hm01 - THRESH_) > 0.0f) ? 1.0f : 0.0f;
        float s10 = ((hm10 - THRESH_) > 0.0f) ? 1.0f : 0.0f;
        float s11 = ((hm11 - THRESH_) > 0.0f) ? 1.0f : 0.0f;

        h_mem[i00] = hm00;  h_mem[i01] = hm01;
        h_mem[i10] = hm10;  h_mem[i11] = hm11;
        s_nat[i00] = s00;   s_nat[i01] = s01;
        s_nat[i10] = s10;   s_nat[i11] = s11;

        scat_spike(wtile0, k0, m_loc, s00);
        scat_spike(wtile0, k1, m_loc, s01);
        scat_spike(wtile1, k0, m_loc, s10);
        scat_spike(wtile1, k1, m_loc, s11);
    }
}

// Output LIF head for one step: o = spikes[b,:] . W_ho[o,:]  (B*O = 5120 dots)
__global__ void step_out(const float* __restrict__ s_nat, const float* __restrict__ W_ho,
                         const float* __restrict__ b_ho, const float* __restrict__ tau_o,
                         float* __restrict__ o_mem, float* __restrict__ o_spk,
                         float* __restrict__ o_sum) {
    int idx = blockIdx.x * blockDim.x + threadIdx.x;
    if (idx >= B_ * O_) return;
    int b = idx / O_, o = idx % O_;
    float alpha = __expf(-1.0f / tau_o[0]);
    const float* s = s_nat + (size_t)b * H_;
    const float* w = W_ho + (size_t)o * H_;
    float sum = 0.0f;
#pragma unroll 4
    for (int k = 0; k < H_; k += 4) {
        sum += s[k] * w[k] + s[k + 1] * w[k + 1] + s[k + 2] * w[k + 2] + s[k + 3] * w[k + 3];
    }
    float osp = o_spk[idx];
    float om  = o_mem[idx] * alpha * (1.0f - osp) + sum + b_ho[o];
    float ns  = ((om - THRESH_) > 0.0f) ? 1.0f : 0.0f;
    o_mem[idx] = om;
    o_spk[idx] = ns;
    o_sum[idx] += ns;
}

__global__ void finalize(const float* __restrict__ o_sum, float* __restrict__ out) {
    int idx = blockIdx.x * blockDim.x + threadIdx.x;
    if (idx < B_ * O_) out[idx] = o_sum[idx] * (1.0f / (float)T_);
}

extern "C" void kernel_launch(void* const* d_in, const int* in_sizes, int n_in,
                              void* d_out, int out_size, void* d_ws, size_t ws_size,
                              hipStream_t stream) {
    (void)in_sizes; (void)n_in; (void)out_size; (void)ws_size;
    const float* input = (const float*)d_in[0];
    const float* W_ih  = (const float*)d_in[1];
    const float* b_ih  = (const float*)d_in[2];
    const float* W_hh  = (const float*)d_in[3];
    const float* b_hh  = (const float*)d_in[4];
    const float* W_ho  = (const float*)d_in[5];
    const float* b_ho  = (const float*)d_in[6];
    const float* tau_h = (const float*)d_in[7];
    const float* tau_o = (const float*)d_in[8];

    char* ws = (char*)d_ws;
    _Float16* a_in   = (_Float16*)(ws + OFF_AIN);
    _Float16* wih_sw = (_Float16*)(ws + OFF_WIH);
    _Float16* whh_sw = (_Float16*)(ws + OFF_WHH);
    float*    xproj  = (float*)(ws + OFF_XPROJ);
    float*    h_mem  = (float*)(ws + OFF_HMEM);
    float*    s_nat  = (float*)(ws + OFF_SNAT);
    _Float16* ssw0   = (_Float16*)(ws + OFF_SSW0);
    _Float16* ssw1   = (_Float16*)(ws + OFF_SSW1);
    float*    o_mem  = (float*)(ws + OFF_OMEM);
    float*    o_spk  = (float*)(ws + OFF_OSPK);
    float*    o_sum  = (float*)(ws + OFF_OSUM);
    float*    out    = (float*)d_out;

    // 1) zero recurrent state (h_mem, spikes (both layouts), output state)
    {
        size_t n = ZERO_BYTES / 4;
        zero_ws<<<(unsigned)((n + 255) / 256), 256, 0, stream>>>(
            (unsigned int*)(ws + OFF_HMEM), n);
    }
    // 2) one-time f16 swizzles (input -> A layout, weights -> B layout)
    {
        int th = (MT / 16) * KT_IN * 32;
        swz_a<<<(th + 255) / 256, 256, 0, stream>>>(input, a_in, MT, NIN, KT_IN);
    }
    {
        int th = (H_ / 16) * KT_IN * 32;
        swz_b<<<(th + 255) / 256, 256, 0, stream>>>(W_ih, wih_sw, H_, NIN, KT_IN);
    }
    {
        int th = (H_ / 16) * KT_HH * 32;
        swz_b<<<(th + 255) / 256, 256, 0, stream>>>(W_hh, whh_sw, H_, H_, KT_HH);
    }
    // 3) input projection GEMM (WMMA, 2x2 blocked), bias folded in
    xproj_gemm<<<(MT / 32) * (H_ / 32) / 8, 256, 0, stream>>>(a_in, wih_sw, b_ih, xproj);

    // 4) serial scan: per step one WMMA recurrence kernel + tiny output head
    for (int t = 0; t < T_; ++t) {
        const _Float16* rd = (t & 1) ? ssw1 : ssw0;
        _Float16*       wr = (t & 1) ? ssw0 : ssw1;
        step_hidden<<<64, 256, 0, stream>>>(whh_sw, rd, wr, h_mem, s_nat,
                                            xproj + (size_t)t * B_ * H_, b_hh, tau_h);
        step_out<<<(B_ * O_ + 255) / 256, 256, 0, stream>>>(s_nat, W_ho, b_ho, tau_o,
                                                            o_mem, o_spk, o_sum);
    }
    // 5) o_sum / T
    finalize<<<(B_ * O_ + 255) / 256, 256, 0, stream>>>(o_sum, out);
}
